// DagLinkExtractor_75935021793678
// MI455X (gfx1250) — compile-verified, matrix operand order checked by reading, not run on hardware
//
#include <hip/hip_runtime.h>
#include <hip/hip_bf16.h>
#include <math.h>

typedef __bf16 v16bf __attribute__((ext_vector_type(16)));
typedef float  v8f   __attribute__((ext_vector_type(8)));

#define B_    8
#define N_    1024
#define HID_  1024
#define NH_   4
#define HD_   256
#define MASKV (-1000000000.0f)

template <bool RA, bool RB>
static __device__ __forceinline__ v8f wmma_bf16(v16bf a, v16bf b, v8f c) {
  // D(16x16,f32) = A(16x32,bf16) * B(32x16,bf16) + C
  return __builtin_amdgcn_wmma_f32_16x16x32_bf16(false, a, false, b, (short)0, c,
                                                 RA, RB);
}

// ---------------- f32 -> bf16 conversion of features ----------------
__global__ void cvt_features_kernel(const float* __restrict__ feat,
                                    __bf16* __restrict__ featB) {
  const int total = B_ * N_ * HID_;
  for (int i = blockIdx.x * blockDim.x + threadIdx.x; i < total;
       i += gridDim.x * blockDim.x) {
    featB[i] = (__bf16)feat[i];
  }
}

// ---------- transpose Wq/Wk to column-major bf16 (WT[n][k] = W[k][n]) ----------
__global__ void transpose_w_kernel(const float* __restrict__ Wq,
                                   const float* __restrict__ Wk,
                                   __bf16* __restrict__ WqT,
                                   __bf16* __restrict__ WkT) {
  const int total = HID_ * HID_;
  for (int idx = blockIdx.x * blockDim.x + threadIdx.x; idx < 2 * total;
       idx += gridDim.x * blockDim.x) {
    int which = idx / total;
    int e = idx - which * total;
    int n = e / HID_;
    int k = e - n * HID_;
    if (which == 0)
      WqT[(size_t)n * HID_ + k] = (__bf16)Wq[(size_t)k * HID_ + n];
    else
      WkT[(size_t)n * HID_ + k] = (__bf16)Wk[(size_t)k * HID_ + n];
  }
}

// ---------------- gates: log_softmax(features @ Wg + bg) ----------------
__global__ void gates_kernel(const float* __restrict__ feat,
                             const float* __restrict__ Wg,
                             const float* __restrict__ bg,
                             float* __restrict__ logGates) {
  int row = blockIdx.x * blockDim.x + threadIdx.x;
  if (row >= B_ * N_) return;
  const float* f = feat + (size_t)row * HID_;
  float a0 = bg[0], a1 = bg[1], a2 = bg[2], a3 = bg[3];
  for (int k = 0; k < HID_; k++) {
    float x = f[k];
    float4 w = *(const float4*)(Wg + (size_t)k * NH_);
    a0 += x * w.x; a1 += x * w.y; a2 += x * w.z; a3 += x * w.w;
  }
  float m = fmaxf(fmaxf(a0, a1), fmaxf(a2, a3));
  float s = __expf(a0 - m) + __expf(a1 - m) + __expf(a2 - m) + __expf(a3 - m);
  float lz = m + __logf(s);
  float4 o = {a0 - lz, a1 - lz, a2 - lz, a3 - lz};
  *(float4*)(logGates + (size_t)row * NH_) = o;
}

// ---------------- per-batch last-valid-index (dead-row predicate) ----------------
__global__ void jmax_kernel(const unsigned char* __restrict__ vm,
                            int* __restrict__ jmax) {
  int b = threadIdx.x;
  if (b >= B_) return;
  int m = -1;
  for (int j = 0; j < N_; j++)
    if (vm[b * N_ + j]) m = j;
  jmax[b] = m;
}

// ---------------- Q/K projection GEMM via WMMA bf16, 32x64 tile per wave ----------------
// grid = (HID/64, B*N/32, 2); one wave per workgroup.
// Each B (weight) fragment feeds two WMMAs (two 16-row A slices) -> halves L2 weight traffic.
__global__ __launch_bounds__(32) void qk_proj_kernel(
    const __bf16* __restrict__ featB,
    const __bf16* __restrict__ WqT, const float* __restrict__ bq, __bf16* __restrict__ Qb,
    const __bf16* __restrict__ WkT, const float* __restrict__ bk, __bf16* __restrict__ Kb) {
  const __bf16* WT  = blockIdx.z ? WkT : WqT;
  const float* bias = blockIdx.z ? bk : bq;
  __bf16* out       = blockIdx.z ? Kb : Qb;

  const int l = threadIdx.x;
  const int i0 = blockIdx.y * 32;   // 32-row tile in flattened (B*N)
  const int n0 = blockIdx.x * 64;   // 64-col output tile
  const int col = l & 15;
  const int khalf = l >> 4;

  v8f accA[4], accB[4];
#pragma unroll
  for (int t = 0; t < 4; t++) { accA[t] = {}; accB[t] = {}; }

  // software-pipelined k-loop: prefetch A for next k-step, B for next t.
  v16bf a0 = *(const v16bf*)(featB + (size_t)(i0 + col) * HID_ + khalf * 16);
  v16bf a1 = *(const v16bf*)(featB + (size_t)(i0 + 16 + col) * HID_ + khalf * 16);
  for (int kk = 0; kk < HID_; kk += 32) {
    const int kn = (kk + 32 < HID_) ? kk + 32 : kk;  // clamped prefetch (in-bounds)
    v16bf a0n = *(const v16bf*)(featB + (size_t)(i0 + col) * HID_ + kn + khalf * 16);
    v16bf a1n = *(const v16bf*)(featB + (size_t)(i0 + 16 + col) * HID_ + kn + khalf * 16);
    v16bf bc = *(const v16bf*)(WT + (size_t)(n0 + col) * HID_ + kk + khalf * 16);
#pragma unroll
    for (int t = 0; t < 4; t++) {
      v16bf bn;
      if (t < 3)
        bn = *(const v16bf*)(WT + (size_t)(n0 + (t + 1) * 16 + col) * HID_ + kk + khalf * 16);
      accA[t] = wmma_bf16<false, true>(a0, bc, accA[t]);   // cache B for next WMMA
      accB[t] = wmma_bf16<false, false>(a1, bc, accB[t]);
      bc = bn;
    }
    a0 = a0n;
    a1 = a1n;
  }
#pragma unroll
  for (int t = 0; t < 4; t++) {
    int n = n0 + t * 16 + col;
    float bv = bias[n];
#pragma unroll
    for (int r = 0; r < 8; r++) {
      int row = r + khalf * 8;
      out[(size_t)(i0 + row) * HID_ + n]      = (__bf16)(accA[t][r] + bv);
      out[(size_t)(i0 + 16 + row) * HID_ + n] = (__bf16)(accB[t][r] + bv);
    }
  }
}

// ---------------- fused scores + masked log-softmax + head mixture ----------------
// grid = (B*N/32); block = 128 (wave w == head w); 32-row i-tile per WG.
// Each wave keeps 32 Q rows of its head resident (16 A fragments); every K fragment
// feeds two WMMAs -> halves L2 K traffic. Inner loop software-pipelined.
__global__ __launch_bounds__(128) void attn_kernel(
    const __bf16* __restrict__ Qb, const __bf16* __restrict__ Kb,
    const unsigned char* __restrict__ vm, const float* __restrict__ logGates,
    const int* __restrict__ jmax, float* __restrict__ out) {
  __shared__ float scratch[NH_][32][17];
  __shared__ float rowMax[NH_][32];
  __shared__ float rowSum[NH_][32];
  __shared__ float alpha[NH_][32];

  const int tid = threadIdx.x;
  const int w = tid >> 5;   // head
  const int l = tid & 31;   // lane
  const int b  = blockIdx.x / (N_ / 32);
  const int i0 = (blockIdx.x % (N_ / 32)) * 32;
  const float rscale = 0.0625f;  // 1/sqrt(HD)
  const int col = l & 15, khalf = l >> 4;

  const __bf16* qrow0 = Qb + (size_t)(b * N_ + i0 + col) * HID_ + w * HD_ + khalf * 16;
  const __bf16* qrow1 = qrow0 + (size_t)16 * HID_;
  const __bf16* kbase = Kb + (size_t)(b * N_) * HID_ + w * HD_ + khalf * 16;

  // preload 32 Q rows of head w: 16 A fragments resident in VGPRs
  v16bf aF0[8], aF1[8];
#pragma unroll
  for (int s8 = 0; s8 < 8; s8++) {
    aF0[s8] = *(const v16bf*)(qrow0 + s8 * 32);
    aF1[s8] = *(const v16bf*)(qrow1 + s8 * 32);
  }

  rowMax[w][l] = -3.0e38f;
  rowSum[w][l] = 0.0f;
  __syncthreads();

  const unsigned char* vmb = vm + b * N_;
  const int jm = jmax[b];

  // ---- pass 1: online max / sum-exp over j ----
  for (int jt = 0; jt < N_ / 16; jt++) {
    const int j0 = jt * 16;
    // prefetch next tile's K row for this lane into caches
    if (j0 + 16 < N_)
      __builtin_prefetch(kbase + (size_t)(j0 + 16 + col) * HID_, 0, 3);
    v8f acc0 = {}, acc1 = {};
    v16bf bc = *(const v16bf*)(kbase + (size_t)(j0 + col) * HID_);
#pragma unroll
    for (int s8 = 0; s8 < 8; s8++) {
      v16bf bn;
      if (s8 < 7) bn = *(const v16bf*)(kbase + (size_t)(j0 + col) * HID_ + (s8 + 1) * 32);
      acc0 = wmma_bf16<false, true>(aF0[s8], bc, acc0);   // cache B for next WMMA
      acc1 = wmma_bf16<false, false>(aF1[s8], bc, acc1);
      bc = bn;
    }
#pragma unroll
    for (int r = 0; r < 8; r++) {
      int row0 = r + khalf * 8;
      int row1 = row0 + 16;
      int j = j0 + col;
      bool vmj = vm != nullptr && vmb[j];
      {
        int i = i0 + row0;
        bool dead = (jm <= i);
        bool v = ((j > i) && vmj) || dead;
        scratch[w][row0][col] = v ? acc0[r] * rscale : MASKV;
      }
      {
        int i = i0 + row1;
        bool dead = (jm <= i);
        bool v = ((j > i) && vmj) || dead;
        scratch[w][row1][col] = v ? acc1[r] * rscale : MASKV;
      }
    }
    __syncthreads();
    {
      float m_old = rowMax[w][l], s_old = rowSum[w][l];
      float tmax = MASKV;
#pragma unroll
      for (int c = 0; c < 16; c++) tmax = fmaxf(tmax, scratch[w][l][c]);
      float m_new = fmaxf(m_old, tmax);
      float s = s_old * __expf(m_old - m_new);
#pragma unroll
      for (int c = 0; c < 16; c++) s += __expf(scratch[w][l][c] - m_new);
      rowMax[w][l] = m_new;
      rowSum[w][l] = s;
    }
    __syncthreads();
  }

  {
    float logZ = rowMax[w][l] + __logf(fmaxf(rowSum[w][l], 1e-37f));
    alpha[w][l] = logGates[(size_t)(b * N_ + i0 + l) * NH_ + w] - logZ;
  }
  __syncthreads();

  // ---- pass 2: recompute scores, mix heads, write log-links ----
  for (int jt = 0; jt < N_ / 16; jt++) {
    const int j0 = jt * 16;
    if (j0 + 16 < N_)
      __builtin_prefetch(kbase + (size_t)(j0 + 16 + col) * HID_, 0, 3);
    v8f acc0 = {}, acc1 = {};
    v16bf bc = *(const v16bf*)(kbase + (size_t)(j0 + col) * HID_);
#pragma unroll
    for (int s8 = 0; s8 < 8; s8++) {
      v16bf bn;
      if (s8 < 7) bn = *(const v16bf*)(kbase + (size_t)(j0 + col) * HID_ + (s8 + 1) * 32);
      acc0 = wmma_bf16<false, true>(aF0[s8], bc, acc0);
      acc1 = wmma_bf16<false, false>(aF1[s8], bc, acc1);
      bc = bn;
    }
#pragma unroll
    for (int r = 0; r < 8; r++) {
      int row0 = r + khalf * 8;
      int row1 = row0 + 16;
      int j = j0 + col;
      bool vmj = vmb[j] != 0;
      {
        int i = i0 + row0;
        bool dead = (jm <= i);
        bool v = ((j > i) && vmj) || dead;
        float sc = v ? acc0[r] * rscale : MASKV;
        scratch[w][row0][col] = sc + alpha[w][row0];
      }
      {
        int i = i0 + row1;
        bool dead = (jm <= i);
        bool v = ((j > i) && vmj) || dead;
        float sc = v ? acc1[r] * rscale : MASKV;
        scratch[w][row1][col] = sc + alpha[w][row1];
      }
    }
    __syncthreads();
#pragma unroll
    for (int e = tid; e < 512; e += 128) {
      int row = e >> 4, c = e & 15;
      float sum = __expf(scratch[0][row][c]) + __expf(scratch[1][row][c]) +
                  __expf(scratch[2][row][c]) + __expf(scratch[3][row][c]);
      int i = i0 + row;
      bool dead = (jm <= i);
      float o = (dead || !(sum > 0.0f)) ? MASKV : __logf(sum);
      out[(size_t)b * N_ * N_ + (size_t)i * N_ + (j0 + c)] = o;
    }
    __syncthreads();
  }
}

// ---------------- host launcher ----------------
extern "C" void kernel_launch(void* const* d_in, const int* in_sizes, int n_in,
                              void* d_out, int out_size, void* d_ws, size_t ws_size,
                              hipStream_t stream) {
  (void)in_sizes; (void)n_in; (void)out_size; (void)ws_size;
  const float*         features   = (const float*)d_in[0];
  const unsigned char* valid_mask = (const unsigned char*)d_in[1];  // jnp bool = 1 byte
  const float*         Wq         = (const float*)d_in[2];
  const float*         bq         = (const float*)d_in[3];
  const float*         Wk         = (const float*)d_in[4];
  const float*         bk         = (const float*)d_in[5];
  const float*         Wg         = (const float*)d_in[6];
  const float*         bg         = (const float*)d_in[7];
  float*               out        = (float*)d_out;

  // workspace carve-out (256B aligned slices)
  char* ws = (char*)d_ws;
  size_t off = 0;
  auto carve = [&](size_t bytes) -> void* {
    void* p = ws + off;
    off += (bytes + 255) & ~(size_t)255;
    return p;
  };
  __bf16* featB    = (__bf16*)carve(sizeof(__bf16) * (size_t)B_ * N_ * HID_);
  __bf16* WqT      = (__bf16*)carve(sizeof(__bf16) * (size_t)HID_ * HID_);
  __bf16* WkT      = (__bf16*)carve(sizeof(__bf16) * (size_t)HID_ * HID_);
  __bf16* Qb       = (__bf16*)carve(sizeof(__bf16) * (size_t)B_ * N_ * HID_);
  __bf16* Kb       = (__bf16*)carve(sizeof(__bf16) * (size_t)B_ * N_ * HID_);
  float*  logGates = (float*)carve(sizeof(float) * (size_t)B_ * N_ * NH_);
  int*    jmaxArr  = (int*)carve(sizeof(int) * B_);

  cvt_features_kernel<<<2048, 256, 0, stream>>>(features, featB);
  transpose_w_kernel<<<2048, 256, 0, stream>>>(Wq, Wk, WqT, WkT);
  gates_kernel<<<(B_ * N_) / 128, 128, 0, stream>>>(features, Wg, bg, logGates);
  jmax_kernel<<<1, 8, 0, stream>>>(valid_mask, jmaxArr);
  qk_proj_kernel<<<dim3(HID_ / 64, (B_ * N_) / 32, 2), 32, 0, stream>>>(
      featB, WqT, bq, Qb, WkT, bk, Kb);
  attn_kernel<<<(B_ * N_) / 32, 128, 0, stream>>>(Qb, Kb, valid_mask, logGates,
                                                  jmaxArr, out);
}
